// ChessNetwork_43774306681109
// MI455X (gfx1250) — matrix operand order repaired and checked by reading, not compiled, not run on hardware
//
#include <hip/hip_runtime.h>
#include <math.h>

typedef __attribute__((ext_vector_type(16))) __bf16 v16bf;
typedef __attribute__((ext_vector_type(8)))  float  v8f;

union ABfrag { v16bf v; uint4 u[2]; };

// ---------------------------------------------------------------------------
// Weight repack: res_w (49 layers)[co][ci][ky][kx] fp32 -> [layer][tap][co][ci] bf16
// ---------------------------------------------------------------------------
__global__ void __launch_bounds__(256) cvt_res_w_k(const float* __restrict__ src,
                                                   __bf16* __restrict__ dst) {
    long i = (long)blockIdx.x * 256 + threadIdx.x;
    const long total = 49L * 49 * 64 * 64;
    if (i >= total) return;
    long t = i;
    int ci = t & 63; t >>= 6;
    int co = t & 63; t >>= 6;
    int tap = (int)(t % 49);
    int layer = (int)(t / 49);
    float v = src[(((long)layer * 64 + co) * 64 + ci) * 49 + tap];
    dst[i] = (__bf16)v;
}

__global__ void __launch_bounds__(256) cvt_last_w_k(const float* __restrict__ src,
                                                    __bf16* __restrict__ dst) {
    int i = blockIdx.x * 256 + threadIdx.x;
    if (i >= 7 * 64 * 64) return;
    dst[i] = (__bf16)src[i];   // [s][co][ci] already matches
}

// ---------------------------------------------------------------------------
// First conv: x (N,13,8,8) fp32 -> conv_feat bf16 [n][64 pix][64 ch]
// ch 0..62 = relu(bn(conv7x7)), ch 63 = piece-id plane (exact in bf16)
// ---------------------------------------------------------------------------
__global__ void __launch_bounds__(256) first_conv_k(const float* __restrict__ x,
                                                    const float* __restrict__ w,
                                                    const float* __restrict__ s,
                                                    const float* __restrict__ b,
                                                    __bf16* __restrict__ out) {
    __shared__ float xs[13 * 64];
    const int n = blockIdx.x, tid = threadIdx.x;
    const float* xp = x + (size_t)n * 13 * 64;
    for (int i = tid; i < 13 * 64; i += 256) xs[i] = xp[i];
    __syncthreads();
    for (int idx = tid; idx < 4096; idx += 256) {
        int p = idx >> 6, c = idx & 63;
        float v;
        if (c == 63) {
            v = xs[12 * 64 + p];
        } else {
            int y = p >> 3, xx = p & 7;
            float acc = 0.f;
            for (int ci = 0; ci < 12; ++ci)
                for (int ky = 0; ky < 7; ++ky) {
                    int yy = y + ky - 3;
                    if ((unsigned)yy >= 8u) continue;
                    for (int kx = 0; kx < 7; ++kx) {
                        int xx2 = xx + kx - 3;
                        if ((unsigned)xx2 >= 8u) continue;
                        acc += xs[ci * 64 + yy * 8 + xx2] *
                               w[((c * 12 + ci) * 7 + ky) * 7 + kx];
                    }
                }
            v = acc * s[c] + b[c];
            v = v > 0.f ? v : 0.f;
        }
        out[(size_t)n * 4096 + idx] = (__bf16)v;
    }
}

// ---------------------------------------------------------------------------
// Piece gather: first position per piece id, gather 64 channels, pid history
// ---------------------------------------------------------------------------
__global__ void __launch_bounds__(32) piece_gather_k(const __bf16* __restrict__ feat,
                                                     float* __restrict__ pv,
                                                     float* __restrict__ pid_out) {
    const int n = blockIdx.x, piece = threadIdx.x; // 32 threads
    const __bf16* f = feat + (size_t)n * 4096;
    int idx = 0; bool present = false;
    for (int p = 0; p < 64; ++p) {
        int id = (int)(float)f[p * 64 + 63];
        if (id == piece + 1) { idx = p; present = true; break; }
    }
    float* o = pv + ((size_t)n * 32 + piece) * 64;
    for (int c = 0; c < 64; ++c)
        o[c] = present ? (float)f[idx * 64 + c] : 0.f;
    float pid = present ? (float)(piece + 1) : 0.f;
    for (int t = 0; t < 8; ++t)
        pid_out[((size_t)n * 8 + t) * 32 + piece] = pid;
}

// ---------------------------------------------------------------------------
// WMMA conv kernel. One workgroup = 8 waves = 2 images.
//   in   : bf16 [n][64 pix][64 ch]
//   wT   : bf16 [ntaps][64 co][64 ci]   (ntaps = 49 for 7x7, 1 for 1x1)
//   out  = relu(acc*scale[co] + bias[co] (+ resid))  as bf16 [n][64 pix][64 ch]
// Per tap: 64x64x64 GEMM per image via v_wmma_f32_16x16x32_bf16.
// Weight tap tiles are double-buffered in LDS, prefetched with
// global_load_async_to_lds_b128 (ASYNCcnt) while the current tap computes.
// LDS pitch = 72 bf16 (36 dwords) -> conflict-free ds_load_b128 fragments.
// ---------------------------------------------------------------------------
#define PITCH 72
__global__ void __launch_bounds__(256) conv_wmma_k(const __bf16* __restrict__ in,
                                                   const __bf16* __restrict__ wT,
                                                   const float* __restrict__ scale,
                                                   const float* __restrict__ bias,
                                                   const __bf16* __restrict__ resid,
                                                   __bf16* __restrict__ out,
                                                   int ntaps) {
    __shared__ __align__(16) __bf16 act[2 * 64 * PITCH];   // 2 images
    __shared__ __align__(16) __bf16 wbuf[2 * 64 * PITCH];  // double-buffered tap

    const int n0 = blockIdx.x * 2;
    const int tid = threadIdx.x;

    // ---- async load both activation tiles (1024 x 16B chunks) ----
    #pragma unroll
    for (int i = 0; i < 4; ++i) {
        int q = tid + 256 * i;          // global 16B-chunk id
        int im = q >> 9;                // 512 chunks per image
        int qi = q & 511;
        int p = qi >> 3, r = qi & 7;
        unsigned lds = (unsigned)(size_t)(act + im * 64 * PITCH + p * PITCH + r * 8);
        unsigned long long ga =
            (unsigned long long)(size_t)(in + (size_t)(n0 + im) * 4096) + (size_t)qi * 16;
        asm volatile("global_load_async_to_lds_b128 %0, %1, off"
                     :: "v"(lds), "v"(ga) : "memory");
    }
    // ---- async load weight tap 0 (512 x 16B chunks) ----
    #pragma unroll
    for (int j = 0; j < 2; ++j) {
        int q = tid * 2 + j;
        int co = q >> 3, r = q & 7;
        unsigned lds = (unsigned)(size_t)(wbuf + co * PITCH + r * 8);
        unsigned long long ga = (unsigned long long)(size_t)wT + (size_t)q * 16;
        asm volatile("global_load_async_to_lds_b128 %0, %1, off"
                     :: "v"(lds), "v"(ga) : "memory");
    }
    asm volatile("s_wait_asynccnt 0" ::: "memory");
    __syncthreads();

    const int wave  = tid >> 5;
    const int lane  = tid & 31;
    const int l15   = lane & 15;
    const int lhalf = lane >> 4;        // selects K sub-range per ISA layout
    const int img   = wave >> 2;        // waves 0-3 -> image 0, 4-7 -> image 1
    const int mbase = (wave & 3) * 16;  // 16-pixel M strip
    const __bf16* atile = act + img * 64 * PITCH;

    v8f acc[4] = {v8f{0}, v8f{0}, v8f{0}, v8f{0}};

    const int m  = mbase + l15;
    const int py = m >> 3, px = m & 7;

    for (int tap = 0; tap < ntaps; ++tap) {
        const int cur = tap & 1;
        // prefetch next tap's weights into the other buffer
        if (tap + 1 < ntaps) {
            const __bf16* wnext = wT + (size_t)(tap + 1) * 4096;
            #pragma unroll
            for (int j = 0; j < 2; ++j) {
                int q = tid * 2 + j;
                int co = q >> 3, r = q & 7;
                unsigned lds = (unsigned)(size_t)(wbuf + (cur ^ 1) * 64 * PITCH +
                                                  co * PITCH + r * 8);
                unsigned long long ga =
                    (unsigned long long)(size_t)wnext + (size_t)q * 16;
                asm volatile("global_load_async_to_lds_b128 %0, %1, off"
                             :: "v"(lds), "v"(ga) : "memory");
            }
        }

        int dy = 0, dx = 0;
        if (ntaps != 1) { dy = tap / 7 - 3; dx = tap % 7 - 3; }
        const int yy = py + dy, xx = px + dx;
        const bool valid = ((unsigned)yy < 8u) && ((unsigned)xx < 8u);
        const int sp = (yy << 3) + xx;
        const __bf16* wl = wbuf + cur * 64 * PITCH;

        #pragma unroll
        for (int ks = 0; ks < 2; ++ks) {
            const int kbase = ks * 32;
            ABfrag a;
            if (valid) {
                // 16-bit A 16x32 layout: lane<16 -> K {0..7,16..23},
                // lane>=16 -> K {8..15,24..31} of this 32-wide k-step
                const uint4* pa =
                    (const uint4*)(atile + sp * PITCH + kbase + lhalf * 8);
                a.u[0] = pa[0];
                a.u[1] = pa[2];
            } else {
                a.u[0] = uint4{0, 0, 0, 0};
                a.u[1] = uint4{0, 0, 0, 0};
            }
            #pragma unroll
            for (int nt = 0; nt < 4; ++nt) {
                const int co = nt * 16 + l15;
                // B 32x16: lane holds column co; lhalf selects K {0..7,16..23} / {8..15,24..31}
                const uint4* pb =
                    (const uint4*)(wl + co * PITCH + kbase + lhalf * 8);
                ABfrag bf;
                bf.u[0] = pb[0];
                bf.u[1] = pb[2];
                acc[nt] = __builtin_amdgcn_wmma_f32_16x16x32_bf16(
                    false, a.v, false, bf.v, (short)0, acc[nt], false, false);
            }
        }
        // drain this wave's prefetch, then WG-wide handoff of buffers
        asm volatile("s_wait_asynccnt 0" ::: "memory");
        __syncthreads();
    }

    // C layout: VGPR i -> row mbase + i + (lane>=16 ? 8 : 0), col = l15
    const int n = n0 + img;
    #pragma unroll
    for (int nt = 0; nt < 4; ++nt) {
        const int co = nt * 16 + l15;
        const float sc = scale[co];
        const float bi = bias[co];
        #pragma unroll
        for (int i = 0; i < 8; ++i) {
            const int mr = mbase + i + lhalf * 8;
            float v = acc[nt][i] * sc + bi;
            if (resid) v += (float)resid[(size_t)n * 4096 + mr * 64 + co];
            v = v > 0.f ? v : 0.f;
            out[(size_t)n * 4096 + mr * 64 + co] = (__bf16)v;
        }
    }
}

// ---------------------------------------------------------------------------
// Value-head 1x1 conv: v0[n][p] = relu(vs * dot(feat[n][p][:], w) + vb)
// ---------------------------------------------------------------------------
__global__ void __launch_bounds__(64) vconv_k(const __bf16* __restrict__ feat,
                                              const float* __restrict__ w,
                                              const float* __restrict__ vs,
                                              const float* __restrict__ vb,
                                              float* __restrict__ v0) {
    const int n = blockIdx.x, p = threadIdx.x;
    const __bf16* f = feat + (size_t)n * 4096 + p * 64;
    float acc = 0.f;
    for (int c = 0; c < 64; ++c) acc += (float)f[c] * w[c];
    float v = acc * vs[0] + vb[0];
    v0[(size_t)n * 64 + p] = v > 0.f ? v : 0.f;
}

// ---------------------------------------------------------------------------
// fp32 tiled GEMM: C[m,n] = act(sum_k A[m,k]*B[n,k] + bias[n])   (B is NxK)
// ---------------------------------------------------------------------------
__global__ void __launch_bounds__(256) gemm_nt_k(const float* __restrict__ A,
                                                 const float* __restrict__ B,
                                                 const float* __restrict__ bias,
                                                 float* __restrict__ C,
                                                 int M, int N, int K, int relu) {
    __shared__ float As[16][16];
    __shared__ float Bs[16][17];
    const int tx = threadIdx.x, ty = threadIdx.y;
    const int row = blockIdx.y * 16 + ty;
    const int col = blockIdx.x * 16 + tx;
    float acc = 0.f;
    for (int k0 = 0; k0 < K; k0 += 16) {
        As[ty][tx] = (row < M && (k0 + tx) < K) ? A[(size_t)row * K + k0 + tx] : 0.f;
        int brow = blockIdx.x * 16 + ty;
        Bs[ty][tx] = (brow < N && (k0 + tx) < K) ? B[(size_t)brow * K + k0 + tx] : 0.f;
        __syncthreads();
        #pragma unroll
        for (int k = 0; k < 16; ++k) acc += As[ty][k] * Bs[tx][k];
        __syncthreads();
    }
    if (row < M && col < N) {
        float v = acc + (bias ? bias[col] : 0.f);
        if (relu) v = v > 0.f ? v : 0.f;
        C[(size_t)row * N + col] = v;
    }
}

// ---------------------------------------------------------------------------
// value = tanh(dot(v1[n], w) + b)
// ---------------------------------------------------------------------------
__global__ void __launch_bounds__(256) value_final_k(const float* __restrict__ v1,
                                                     const float* __restrict__ w,
                                                     const float* __restrict__ b,
                                                     float* __restrict__ out) {
    const int n = blockIdx.x * blockDim.x + threadIdx.x;
    if (n >= 1024) return;
    float acc = 0.f;
    for (int k = 0; k < 256; ++k) acc += v1[(size_t)n * 256 + k] * w[k];
    out[n] = tanhf(acc + b[0]);
}

// ---------------------------------------------------------------------------
extern "C" void kernel_launch(void* const* d_in, const int* in_sizes, int n_in,
                              void* d_out, int out_size, void* d_ws, size_t ws_size,
                              hipStream_t stream) {
    (void)in_sizes; (void)n_in; (void)out_size; (void)ws_size;
    const float* x       = (const float*)d_in[0];
    const float* first_w = (const float*)d_in[1];
    const float* first_s = (const float*)d_in[2];
    const float* first_b = (const float*)d_in[3];
    const float* res_w   = (const float*)d_in[4];
    const float* res_s   = (const float*)d_in[5];
    const float* res_b   = (const float*)d_in[6];
    const float* last_w  = (const float*)d_in[7];
    const float* last_s  = (const float*)d_in[8];
    const float* last_b  = (const float*)d_in[9];
    const float* pfc1_w  = (const float*)d_in[10];
    const float* pfc1_b  = (const float*)d_in[11];
    const float* pfc2_w  = (const float*)d_in[12];
    const float* pfc2_b  = (const float*)d_in[13];
    const float* vconv_w = (const float*)d_in[14];
    const float* v_s     = (const float*)d_in[15];
    const float* v_b     = (const float*)d_in[16];
    const float* vfc1_w  = (const float*)d_in[17];
    const float* vfc1_b  = (const float*)d_in[18];
    const float* vfc2_w  = (const float*)d_in[19];
    const float* vfc2_b  = (const float*)d_in[20];

    float* outp = (float*)d_out;
    const size_t POLICY_OFF = 0;
    const size_t VALUE_OFF  = 1024UL * 1700;
    const size_t PID_OFF    = VALUE_OFF + 1024;

    // workspace carve-up (256-B aligned)
    char* ws = (char*)d_ws;
    size_t off = 0;
    auto carve = [&](size_t bytes) {
        char* p = ws + off;
        off = (off + bytes + 255) & ~(size_t)255;
        return p;
    };
    __bf16* res_wT  = (__bf16*)carve(49UL * 49 * 64 * 64 * 2); // 18.8 MB
    __bf16* last_wT = (__bf16*)carve(7UL * 64 * 64 * 2);
    __bf16* F       = (__bf16*)carve(1024UL * 4096 * 2);       // conv_feat
    __bf16* bufA    = (__bf16*)carve(1024UL * 4096 * 2);
    __bf16* bufB    = (__bf16*)carve(1024UL * 4096 * 2);
    float*  pv      = (float*) carve(1024UL * 32 * 64 * 4);    // piece_vec
    float*  hid     = (float*) carve(1024UL * 256 * 4);
    float*  v0      = (float*) carve(1024UL * 64 * 4);
    float*  v1      = (float*) carve(1024UL * 256 * 4);

    // 1) weight repack fp32 -> bf16
    cvt_res_w_k<<<(49 * 49 * 64 * 64 + 255) / 256, 256, 0, stream>>>(res_w, res_wT);
    cvt_last_w_k<<<(7 * 64 * 64 + 255) / 256, 256, 0, stream>>>(last_w, last_wT);

    // 2) first conv -> F
    first_conv_k<<<1024, 256, 0, stream>>>(x, first_w, first_s, first_b, F);

    // 3) piece gather (reads pre-tower F)
    piece_gather_k<<<1024, 32, 0, stream>>>(F, pv, outp + PID_OFF);

    // 4) residual tower: 7 stages x (7 WMMA convs + WMMA 1x1 with residual)
    for (int s = 0; s < 7; ++s) {
        const __bf16* cur = F;
        __bf16* nxt = bufA;
        for (int j = 0; j < 7; ++j) {
            const int layer = s * 7 + j;
            conv_wmma_k<<<512, 256, 0, stream>>>(
                cur, res_wT + (size_t)layer * 49 * 4096,
                res_s + layer * 64, res_b + layer * 64,
                nullptr, nxt, 49);
            cur = nxt;
            nxt = (cur == bufA) ? bufB : bufA;
        }
        conv_wmma_k<<<512, 256, 0, stream>>>(
            cur, last_wT + (size_t)s * 4096,
            last_s + s * 64, last_b + s * 64,
            F, F, 1);
    }

    // 5) value-head 1x1 conv
    vconv_k<<<1024, 64, 0, stream>>>(F, vconv_w, v_s, v_b, v0);

    // 6) policy head FCs
    {
        dim3 blk(16, 16);
        dim3 g1((256 + 15) / 16, (1024 + 15) / 16);
        gemm_nt_k<<<g1, blk, 0, stream>>>(pv, pfc1_w, pfc1_b, hid, 1024, 256, 2048, 1);
        dim3 g2((1700 + 15) / 16, (1024 + 15) / 16);
        gemm_nt_k<<<g2, blk, 0, stream>>>(hid, pfc2_w, pfc2_b, outp + POLICY_OFF,
                                          1024, 1700, 256, 0);
        dim3 g3((256 + 15) / 16, (1024 + 15) / 16);
        gemm_nt_k<<<g3, blk, 0, stream>>>(v0, vfc1_w, vfc1_b, v1, 1024, 256, 64, 1);
    }

    // 7) value finisher
    value_final_k<<<4, 256, 0, stream>>>(v1, vfc2_w, vfc2_b, outp + VALUE_OFF);
}
